// ModulatedDeformConvWithOffset_13202729468339
// MI455X (gfx1250) — matrix operand order repaired and checked by reading, not compiled, not run on hardware
//
#include <hip/hip_runtime.h>

// ---------------------------------------------------------------------------
// DCNv2 forward for MI455X (gfx1250): bf16 WMMA implicit-GEMM, fused gather.
// Rev 2: branch-free padded loads (no exec-mask juggling), 128-pixel tiles
// (4 WMMAs/wave/barrier, 16-channel reuse of each bilinear coordinate).
// ---------------------------------------------------------------------------

typedef __bf16 v16bf __attribute__((ext_vector_type(16)));
typedef float  v8f   __attribute__((ext_vector_type(8)));

struct U8 { uint4 a, b; };   // 32-byte, 16-byte-aligned fragment carrier

#define Hh      128
#define Ww      128
#define Cc      64
#define COo     64
#define HW      16384          // H*W
#define KTOT    576            // C * 9
#define NCHUNK  18             // KTOT / 32
#define PIXT    128            // pixels per workgroup tile (one full row)
#define BSTRIDE 20             // dwords per pixel row in lds_b (64B data + 16B pad)

static __device__ __forceinline__ unsigned short f2bf(float f) {
  unsigned int u = __float_as_uint(f);
  u += 0x7FFFu + ((u >> 16) & 1u);           // round-to-nearest-even
  return (unsigned short)(u >> 16);
}
static __device__ __forceinline__ unsigned int packbf(float lo, float hi) {
  return (unsigned int)f2bf(lo) | ((unsigned int)f2bf(hi) << 16);
}
static __device__ __forceinline__ v16bf ldfrag(const unsigned int* p) {
  U8 t; t.a = ((const uint4*)p)[0]; t.b = ((const uint4*)p)[1];
  return __builtin_bit_cast(v16bf, t);
}
static __device__ __forceinline__ int clamp127(int v) {
  return v < 0 ? 0 : (v > 127 ? 127 : v);
}

// ---------------------------------------------------------------------------
// Kernel 0: repack weights into WMMA A-fragment order (bf16).
// frag[(kc*nCB + cb)*256 + lane*8 + v] ; K order: k = tap*64 + c.
// ---------------------------------------------------------------------------
__global__ __launch_bounds__(256) void dcn_prep(
    const float* __restrict__ w_dcn, const float* __restrict__ w_off,
    unsigned int* __restrict__ frag_dcn, unsigned int* __restrict__ frag_off)
{
  int g = blockIdx.x * 256 + threadIdx.x;          // 27648 total = 18432 + 9216
  const bool isdcn = g < 18432;
  int gg  = isdcn ? g : g - 18432;
  const int nCB = isdcn ? 4 : 2;
  int kc  = gg / (nCB * 256);
  int rem = gg % (nCB * 256);
  int cb  = rem >> 8;
  int idx = rem & 255;
  int lane = idx >> 3, v = idx & 7;
  int half = lane >> 4, mrow = lane & 15;
  int row  = cb * 16 + mrow;
  unsigned int o = 0;
#pragma unroll
  for (int j = 0; j < 2; ++j) {
    int i  = v * 2 + j;
    int Kl = i + half * 16;                        // A-matrix 16x32 bf16 layout (contiguous k = 16*half + i)
    int k  = kc * 32 + Kl;
    int tap = k >> 6, c = k & 63;
    int ty = tap / 3, tx = tap % 3;
    float val;
    if (isdcn) val = w_dcn[((row * Cc + c) * 3 + ty) * 3 + tx];
    else       val = (row < 27) ? w_off[((row * Cc + c) * 3 + ty) * 3 + tx] : 0.f;
    o |= (unsigned int)f2bf(val) << (16 * j);
  }
  (isdcn ? frag_dcn : frag_off)[(kc * nCB + cb) * 256 + idx] = o;
}

// ---------------------------------------------------------------------------
// Kernel 1: offset conv (regular 3x3 im2col GEMM) -> om[B][27][H][W].
// 32(row-padded) x 128-pixel tile; 8 waves x 2 C-tiles of 16x16.
// ---------------------------------------------------------------------------
__global__ __launch_bounds__(256) void dcn_offset_conv(
    const float* __restrict__ x, const float* __restrict__ b_off,
    const unsigned int* __restrict__ frag_off, float* __restrict__ om)
{
  __shared__ unsigned int lds_b[2][PIXT * BSTRIDE];   // 20480 B
  const int t = threadIdx.x;
  const int lane = t & 31, wv = t >> 5;
  const int h  = blockIdx.y;
  const int bb = blockIdx.z;
  const int cbw = wv & 1;                // Co block (0..1)
  const int nb0 = (wv >> 1) << 5;        // pixel sub-block pair base
  const int pix = t & 127;
  const int cby = (t >> 7) << 4;         // 16 channels per thread
  v8f acc0 = {}, acc1 = {};

  for (int kc = 0; kc < NCHUNK; ++kc) {
    const int buf = kc & 1;
    const int tap = kc >> 1;
    const int c0  = (kc & 1) << 5;
    const int ty = tap / 3, tx = tap % 3;
    const int y  = h + ty - 1;
    const int xw = pix + tx - 1;
    // branch-free zero padding: clamped address, 0/1 multiplier
    const float pad = (((unsigned)y < (unsigned)Hh) && ((unsigned)xw < (unsigned)Ww))
                          ? 1.f : 0.f;
    const float* xc = x + ((size_t)(bb * Cc + c0 + cby) * Hh + clamp127(y)) * Ww
                        + clamp127(xw);
    unsigned int* dstp = &lds_b[buf][pix * BSTRIDE + (cby >> 1)];
#pragma unroll
    for (int q = 0; q < 8; ++q) {
      float a  = xc[0] * pad; xc += HW;
      float bv = xc[0] * pad; xc += HW;
      dstp[q] = packbf(a, bv);
    }
    __syncthreads();
    v16bf afrag = ldfrag(frag_off + ((kc * 2 + cbw) * 256 + lane * 8));
    v16bf b0 = ldfrag(&lds_b[buf][(nb0      + (lane & 15)) * BSTRIDE + ((lane >> 4) << 3)]);
    v16bf b1 = ldfrag(&lds_b[buf][(nb0 + 16 + (lane & 15)) * BSTRIDE + ((lane >> 4) << 3)]);
    acc0 = __builtin_amdgcn_wmma_f32_16x16x32_bf16(false, afrag, false, b0,
                                                   (short)0, acc0, false, false);
    acc1 = __builtin_amdgcn_wmma_f32_16x16x32_bf16(false, afrag, false, b1,
                                                   (short)0, acc1, false, false);
  }
  const int half = lane >> 4;
  const int n = nb0 + (lane & 15);
#pragma unroll
  for (int r = 0; r < 8; ++r) {
    int co = cbw * 16 + r + half * 8;
    if (co < 27) {
      size_t rowb = ((size_t)(bb * 27 + co) * Hh + h) * Ww;
      float bias = b_off[co];
      om[rowb + n]      = acc0[r] + bias;
      om[rowb + n + 16] = acc1[r] + bias;
    }
  }
}

// ---------------------------------------------------------------------------
// Kernel 2: fused deformable gather + 64x128 WMMA GEMM tile -> out.
// 8 waves x 4 C-tiles; A-fragment reused 4x per chunk.
// ---------------------------------------------------------------------------
__global__ __launch_bounds__(256) void dcn_main(
    const float* __restrict__ x, const float* __restrict__ om,
    const unsigned int* __restrict__ frag_dcn, const float* __restrict__ b_dcn,
    float* __restrict__ out)
{
  __shared__ unsigned int lds_b[2][PIXT * BSTRIDE];     // 20480 B
  __shared__ unsigned int lds_crd[9 * PIXT * 8];        // 36864 B: 4 offs + 4 wts
  const int t = threadIdx.x;
  const int lane = t & 31, wv = t >> 5;
  const int h  = blockIdx.y;
  const int bb = blockIdx.z;

  // ---- Phase 1: bilinear corner offsets + mask-folded weights (per tap,pix)
  for (int e = t; e < 9 * PIXT; e += 256) {
    const int tap = e >> 7, pix = e & 127;
    const int ty = tap / 3, tx = tap % 3;
    const size_t sb = (size_t)(bb * 27) * HW + (size_t)h * Ww + pix;
    float oy = om[sb + (size_t)tap * HW];
    float ox = om[sb + (size_t)(9 + tap) * HW];
    float mm = om[sb + (size_t)(18 + tap) * HW];
    float msk = 1.f / (1.f + __expf(-mm));
    float py = oy + (float)(h - 1 + ty);
    float px = ox + (float)(pix - 1 + tx);
    float y0f = floorf(py), x0f = floorf(px);
    float dy = py - y0f, dx = px - x0f;
    int y0 = (int)y0f, x0 = (int)x0f;
    int y1 = y0 + 1,   x1 = x0 + 1;
    float vy0 = ((unsigned)y0 < 128u) ? 1.f : 0.f;
    float vy1 = ((unsigned)y1 < 128u) ? 1.f : 0.f;
    float vx0 = ((unsigned)x0 < 128u) ? 1.f : 0.f;
    float vx1 = ((unsigned)x1 < 128u) ? 1.f : 0.f;
    int cy0 = clamp127(y0), cy1 = clamp127(y1);
    int cx0 = clamp127(x0), cx1 = clamp127(x1);
    unsigned int* cr = &lds_crd[e * 8];
    cr[0] = (unsigned int)(cy0 * Ww + cx0);
    cr[1] = (unsigned int)(cy0 * Ww + cx1);
    cr[2] = (unsigned int)(cy1 * Ww + cx0);
    cr[3] = (unsigned int)(cy1 * Ww + cx1);
    cr[4] = __float_as_uint((1.f - dy) * (1.f - dx) * msk * vy0 * vx0);
    cr[5] = __float_as_uint((1.f - dy) * dx         * msk * vy0 * vx1);
    cr[6] = __float_as_uint(dy * (1.f - dx)         * msk * vy1 * vx0);
    cr[7] = __float_as_uint(dy * dx                 * msk * vy1 * vx1);
  }
  __syncthreads();

  // ---- Phase 2: K-loop (18 x 32-deep bf16 WMMA), gather fused into B tile
  const int cbw = wv & 3;                // Co block (4)
  const int nb0 = (wv >> 2) << 6;        // pixel block base (0 or 64)
  const int pix = t & 127;
  const int cby = (t >> 7) << 4;         // 16 channels per thread
  v8f accv[4] = {};

  for (int kc = 0; kc < NCHUNK; ++kc) {
    const int buf = kc & 1;
    const int tap = kc >> 1;
    const int c0  = (kc & 1) << 5;
    const unsigned int* cr = &lds_crd[((tap << 7) | pix) * 8];
    const int o00 = (int)cr[0], o01 = (int)cr[1], o10 = (int)cr[2], o11 = (int)cr[3];
    const float w00 = __uint_as_float(cr[4]);
    const float w01 = __uint_as_float(cr[5]);
    const float w10 = __uint_as_float(cr[6]);
    const float w11 = __uint_as_float(cr[7]);
    const float* xc = x + (size_t)(bb * Cc + c0 + cby) * HW;   // L2-resident gathers
    unsigned int* dstp = &lds_b[buf][pix * BSTRIDE + (cby >> 1)];
#pragma unroll
    for (int q = 0; q < 8; ++q) {
      float a  = w00 * xc[o00] + w01 * xc[o01] + w10 * xc[o10] + w11 * xc[o11];
      xc += HW;
      float bv = w00 * xc[o00] + w01 * xc[o01] + w10 * xc[o10] + w11 * xc[o11];
      xc += HW;
      dstp[q] = packbf(a, bv);
    }
    __syncthreads();
    if (kc + 1 < NCHUNK)   // global_prefetch next A fragment
      __builtin_prefetch(frag_dcn + (((kc + 1) * 4 + cbw) * 256 + lane * 8), 0, 1);
    v16bf afrag = ldfrag(frag_dcn + ((kc * 4 + cbw) * 256 + lane * 8));
#pragma unroll
    for (int j = 0; j < 4; ++j) {
      v16bf bf = ldfrag(&lds_b[buf][(nb0 + j * 16 + (lane & 15)) * BSTRIDE
                                    + ((lane >> 4) << 3)]);
      accv[j] = __builtin_amdgcn_wmma_f32_16x16x32_bf16(false, afrag, false, bf,
                                                        (short)0, accv[j], false, false);
    }
  }

  // ---- Epilogue: bias + store (C/D layout: M = r + 8*(lane/16), N = lane%16)
  const int half = lane >> 4;
#pragma unroll
  for (int r = 0; r < 8; ++r) {
    int co = cbw * 16 + r + half * 8;
    float bias = b_dcn[co];
    size_t rowb = ((size_t)(bb * COo + co) * Hh + h) * Ww;
#pragma unroll
    for (int j = 0; j < 4; ++j)
      out[rowb + nb0 + j * 16 + (lane & 15)] = accv[j][r] + bias;
  }
}

// ---------------------------------------------------------------------------
extern "C" void kernel_launch(void* const* d_in, const int* in_sizes, int n_in,
                              void* d_out, int out_size, void* d_ws, size_t ws_size,
                              hipStream_t stream)
{
  const float* x     = (const float*)d_in[0];
  const float* w_off = (const float*)d_in[1];
  const float* b_off = (const float*)d_in[2];
  const float* w_dcn = (const float*)d_in[3];
  const float* b_dcn = (const float*)d_in[4];
  float* out = (float*)d_out;

  // workspace: om (14,155,776 B) | frag_dcn (73,728 B) | frag_off (36,864 B)
  float*        om       = (float*)d_ws;
  unsigned int* frag_dcn = (unsigned int*)((char*)d_ws + 14155776);
  unsigned int* frag_off = frag_dcn + 18432;

  dcn_prep<<<108, 256, 0, stream>>>(w_dcn, w_off, frag_dcn, frag_off);
  dim3 grid(Ww / PIXT, Hh, 8);           // 1 x 128 x 8 = 1024 workgroups
  dcn_offset_conv<<<grid, 256, 0, stream>>>(x, b_off, frag_off, om);
  dcn_main<<<grid, 256, 0, stream>>>(x, om, frag_dcn, b_dcn, out);
}
